// SimplifiedMambaBlock_86835648790971
// MI455X (gfx1250) — matrix operand order repaired and checked by reading, not compiled, not run on hardware
//
#include <hip/hip_runtime.h>
#include <cstdint>

// Problem constants (from reference): B=2, S=4096, D=1024, N=16, K_conv=4
#define BB 2
#define SS 4096
#define DD 1024
#define NN 16
#define KCONV 4
#define MM (BB * SS)        // 8192 rows
#define TWO_D (2 * DD)      // 2048
#define NCH 16              // scan chunks per sequence
#define TCH (SS / NCH)      // 256 timesteps per chunk

#define LDSPAD 40           // halves per 32-half row: 16B pad -> conflict-free 16-row col reads

typedef __attribute__((ext_vector_type(16))) __bf16       v16bf;
typedef __attribute__((ext_vector_type(8)))  float        v8f;
typedef __attribute__((ext_vector_type(8)))  unsigned int v8u;
typedef __attribute__((ext_vector_type(4)))  unsigned int v4u;
typedef __attribute__((ext_vector_type(4)))  float        v4f;

// ---------------------------------------------------------------- helpers
__device__ __forceinline__ uint16_t f2bf(float f) {
  uint32_t u = __float_as_uint(f);
  u += 0x7fffu + ((u >> 16) & 1u);   // round-to-nearest-even
  return (uint16_t)(u >> 16);
}

__device__ __forceinline__ v8u cat8(v4u lo, v4u hi) {
  v8u u;
#pragma unroll
  for (int i = 0; i < 4; ++i) { u[i] = lo[i]; u[i + 4] = hi[i]; }
  return u;
}

// Async global->LDS copy, 16B per lane (GLOBAL_LOAD_ASYNC_TO_LDS_B128, ASYNCcnt).
__device__ __forceinline__ void async_copy_b128(uint32_t ldsOff, const void* gaddr) {
  asm volatile("global_load_async_to_lds_b128 %0, %1, off"
               :: "v"(ldsOff), "v"(gaddr)
               : "memory");
}
__device__ __forceinline__ void wait_asynccnt0() {
  asm volatile("s_wait_asynccnt 0x0" ::: "memory");
}

// A-matrix fragment (16x32 bf16) from a row base (32 halves + pad).
// ISA 7.12.2: lanes 0-15 (half=0): v0..3 -> K0..7, v4..7 -> K16..23;
//             lanes 16-31 (half=1): v0..3 -> K8..15, v4..7 -> K24..31.
__device__ __forceinline__ v16bf fragA_row(const uint16_t* rowBase, int half) {
  v4u lo = *(const v4u*)(rowBase + half * 8);
  v4u hi = *(const v4u*)(rowBase + 16 + half * 8);
  return __builtin_bit_cast(v16bf, cat8(lo, hi));
}

// B-matrix fragment (32x16 bf16): lane holds column n=lane%16;
// lanes 0-15 hold K=0..7,16..23; lanes 16-31 hold K=8..15,24..31.
__device__ __forceinline__ v16bf fragB_row(const uint16_t* rowBase, int half) {
  v4u lo = *(const v4u*)(rowBase + half * 8);
  v4u hi = *(const v4u*)(rowBase + 16 + half * 8);
  return __builtin_bit_cast(v16bf, cat8(lo, hi));
}

// ---------------------------------------------------------------- LDS-staged GEMM
// C[M, Nc] = A[M, K] * W[Nc, K]^T, bf16 in / fp32 accum.
// Block: 256 threads = 8 waves, 128x128 C tile; wave: 32(M) x 64(N) (2x4 WMMA tiles).
// Double-buffered async (DMA) staging of 128x32 A and B tiles; one barrier per k-step.
__global__ __launch_bounds__(256) void gemm_lds_bf16_wmma(
    const uint16_t* __restrict__ A, const uint16_t* __restrict__ W,
    float* __restrict__ C, int M, int Nc, int K) {
  __shared__ uint16_t smA[2][128 * LDSPAD];
  __shared__ uint16_t smB[2][128 * LDSPAD];

  const int tid  = threadIdx.x;
  const int lane = tid & 31;
  const int half = lane >> 4;
  const int l16  = lane & 15;
  const int wv   = tid >> 5;       // 0..7
  const int mSub = wv & 3;         // 4 waves along M
  const int nSub = wv >> 2;        // 2 waves along N

  const int nBlocks = Nc / 128;
  const int mBase = (blockIdx.x / nBlocks) * 128;
  const int nBase = (blockIdx.x % nBlocks) * 128;

  // cooperative staging map: thread -> (row 0..127, 16-half chunk 0..1)
  const int sr = tid >> 1;
  const int sc = (tid & 1) << 4;
  const uint16_t* gA = A + (size_t)(mBase + sr) * K + sc;
  const uint16_t* gB = W + (size_t)(nBase + sr) * K + sc;
  const int lofs = sr * LDSPAD + sc;

  // LDS byte offsets for this thread's staging slot (low 32 bits == AS(3) offset)
  const uint32_t ldsA = (uint32_t)(uintptr_t)(&smA[0][0] + lofs);
  const uint32_t ldsB = (uint32_t)(uintptr_t)(&smB[0][0] + lofs);
  const uint32_t bstr = 128u * LDSPAD * 2u;  // bytes per LDS buffer

  const int nk = K / 32;

  // stage tile kt into LDS buffer q (4 async b128 per thread -> 16KB tile pair)
  auto stage = [&](int kt, int q) {
    const uint16_t* ga = gA + (size_t)kt * 32;
    const uint16_t* gb = gB + (size_t)kt * 32;
    const uint32_t la = ldsA + (uint32_t)q * bstr;
    const uint32_t lb = ldsB + (uint32_t)q * bstr;
    async_copy_b128(la,      ga);
    async_copy_b128(la + 16, ga + 8);
    async_copy_b128(lb,      gb);
    async_copy_b128(lb + 16, gb + 8);
  };

  // preload tile 0
  stage(0, 0);
  wait_asynccnt0();
  __syncthreads();

  v8f acc[2][4] = {};

  for (int kt = 0; kt < nk; ++kt) {
    const int p = kt & 1;
    if (kt + 1 < nk) {
      stage(kt + 1, 1 - p);       // DMA next tile while WMMAs run
      if (kt + 2 < nk) {
        __builtin_prefetch(gA + (size_t)(kt + 2) * 32, 0, 1);  // global_prefetch_b8
        __builtin_prefetch(gB + (size_t)(kt + 2) * 32, 0, 1);
      }
    }

    // compute from buffer p
    {
      const uint16_t* bufA = &smA[p][0];
      const uint16_t* bufB = &smB[p][0];
      v16bf bf[4];
#pragma unroll
      for (int nt = 0; nt < 4; ++nt)
        bf[nt] = fragB_row(bufB + (nSub * 64 + nt * 16 + l16) * LDSPAD, half);
#pragma unroll
      for (int mt = 0; mt < 2; ++mt) {
        const v16bf af = fragA_row(bufA + (mSub * 32 + mt * 16 + l16) * LDSPAD, half);
#pragma unroll
        for (int nt = 0; nt < 4; ++nt)
          acc[mt][nt] = __builtin_amdgcn_wmma_f32_16x16x32_bf16(
              false, af, false, bf[nt], (short)0, acc[mt][nt], false, false);
      }
    }

    if (kt + 1 < nk) {
      wait_asynccnt0();           // this wave's DMA done; barrier makes all visible
      __syncthreads();
    }
  }

  // C/D layout: lane -> col = lane%16; row = half*8 + v
#pragma unroll
  for (int mt = 0; mt < 2; ++mt) {
    const int rowBase = mBase + mSub * 32 + mt * 16 + half * 8;
#pragma unroll
    for (int v = 0; v < 8; ++v) {
      float* crow = C + (size_t)(rowBase + v) * Nc + nBase + nSub * 64 + l16;
#pragma unroll
      for (int nt = 0; nt < 4; ++nt) crow[nt * 16] = acc[mt][nt][v];
    }
  }
}

// ---------------------------------------------------------------- direct GEMM (skinny Nc)
__device__ __forceinline__ v16bf load_fragA_g(const uint16_t* p, int lane) {
  v8u u;
  const int half = lane >> 4;
#pragma unroll
  for (int v = 0; v < 8; ++v) {
    const int kb = ((v >> 2) << 4) + (half << 3) + ((v & 3) << 1);
    u[v] = *(const uint32_t*)(p + kb);
  }
  return __builtin_bit_cast(v16bf, u);
}
__device__ __forceinline__ v16bf load_fragB_g(const uint16_t* p, int lane) {
  v8u u;
  const int half = lane >> 4;
#pragma unroll
  for (int v = 0; v < 8; ++v) {
    const int kb = ((v >> 2) << 4) + (half << 3) + ((v & 3) << 1);
    u[v] = *(const uint32_t*)(p + kb);
  }
  return __builtin_bit_cast(v16bf, u);
}

template <int NT>
__global__ __launch_bounds__(256) void gemm_bf16_wmma(
    const uint16_t* __restrict__ A, const uint16_t* __restrict__ W,
    float* __restrict__ C, int M, int Nc, int K) {
  const int lane  = threadIdx.x & 31;
  const int wave  = blockIdx.x * (blockDim.x >> 5) + (threadIdx.x >> 5);
  const int nTiles = Nc / (16 * NT);
  const int mTile  = wave / nTiles;
  const int nBase  = (wave % nTiles) * (16 * NT);
  if (mTile * 16 >= M) return;

  const uint16_t* aRow = A + (size_t)(mTile * 16 + (lane & 15)) * K;
  const uint16_t* bRow = W + (size_t)(nBase + (lane & 15)) * K;

  v8f acc[NT] = {};
  for (int k0 = 0; k0 < K; k0 += 32) {
    const v16bf a = load_fragA_g(aRow + k0, lane);
#pragma unroll
    for (int j = 0; j < NT; ++j) {
      const v16bf b = load_fragB_g(bRow + (size_t)j * 16 * K + k0, lane);
      acc[j] = __builtin_amdgcn_wmma_f32_16x16x32_bf16(
          false, a, false, b, (short)0, acc[j], false, false);
    }
  }
  const int mBase = mTile * 16 + (lane >> 4) * 8;
  const int nCol  = nBase + (lane & 15);
#pragma unroll
  for (int v = 0; v < 8; ++v)
#pragma unroll
    for (int j = 0; j < NT; ++j)
      C[(size_t)(mBase + v) * Nc + nCol + j * 16] = acc[j][v];
}

// ---------------------------------------------------------------- elementwise
__global__ void cvt_bf16x4_kernel(const float* __restrict__ src,
                                  uint16_t* __restrict__ dst, int n4) {
  int i = blockIdx.x * blockDim.x + threadIdx.x;
  if (i >= n4) return;
  v4f f = *(const v4f*)(src + (size_t)i * 4);
  uint16_t o[4];
#pragma unroll
  for (int k = 0; k < 4; ++k) o[k] = f2bf(f[k]);
  *(uint64_t*)(dst + (size_t)i * 4) = *(const uint64_t*)o;
}

__global__ void negexp_kernel(const float* __restrict__ src,
                              float* __restrict__ dst, int n) {
  int i = blockIdx.x * blockDim.x + threadIdx.x;
  if (i < n) dst[i] = -__expf(src[i]);
}

// Depthwise causal conv (left pad K-1) + SiLU. Reads x_in = xz[:, 0:D].
__global__ void conv_silu_kernel(const float* __restrict__ xz,
                                 const float* __restrict__ w,
                                 float* __restrict__ xc,
                                 uint16_t* __restrict__ xcb) {
  int i = blockIdx.x * blockDim.x + threadIdx.x;
  if (i >= MM * DD) return;
  const int d = i % DD;
  const int t = (i / DD) % SS;
  const int b = i / (DD * SS);
  float acc = 0.f;
#pragma unroll
  for (int k = 0; k < KCONV; ++k) {
    const int tt = t + k - (KCONV - 1);
    if (tt >= 0)
      acc += w[d * KCONV + k] * xz[(size_t)(b * SS + tt) * TWO_D + d];
  }
  const float s = acc / (1.f + __expf(-acc));
  xc[i] = s;
  xcb[i] = f2bf(s);
}

// dt = softplus(dt_low @ W_dt^T + b_dt); K=16 so plain VALU FMAs.
__global__ void dtproj_softplus_kernel(const float* __restrict__ dtBC,
                                       const float* __restrict__ Wdt,
                                       const float* __restrict__ bdt,
                                       float* __restrict__ dt) {
  int i = blockIdx.x * blockDim.x + threadIdx.x;
  if (i >= MM * DD) return;
  const int d = i % DD;
  const int m = i / DD;
  float acc = bdt[d];
  const float* row = dtBC + (size_t)m * (2 * NN);
  const float* wr  = Wdt + d * NN;
#pragma unroll
  for (int n = 0; n < NN; ++n) acc += row[n] * wr[n];
  dt[i] = (acc > 20.f) ? acc : __logf(1.f + __expf(acc));
}

// ---------------------------------------------------------------- chunked scan
__global__ void scan_phase1_kernel(const float* __restrict__ dt,
                                   const float* __restrict__ dtBC,
                                   const float* __restrict__ xc,
                                   const float* __restrict__ A,
                                   float* __restrict__ Pws,
                                   float* __restrict__ Hloc) {
  int idx = blockIdx.x * blockDim.x + threadIdx.x;
  if (idx >= BB * DD * NCH) return;
  const int d = idx % DD;
  const int c = (idx / DD) % NCH;
  const int b = idx / (DD * NCH);
  float h[NN] = {};
  float P[NN], Ad[NN];
#pragma unroll
  for (int n = 0; n < NN; ++n) { P[n] = 1.f; Ad[n] = A[d * NN + n]; }
  const int t0 = c * TCH;
  for (int t = t0; t < t0 + TCH; ++t) {
    const size_t row = (size_t)b * SS + t;
    const float dtv = dt[row * DD + d];
    const float xv  = xc[row * DD + d];
    const float* Bt = dtBC + row * (2 * NN) + NN;
#pragma unroll
    for (int n = 0; n < NN; ++n) {
      const float e = __expf(Ad[n] * dtv);
      h[n] = h[n] * e + Bt[n] * xv;
      P[n] *= e;
    }
  }
  const size_t o = (size_t)((b * DD + d) * NCH + c) * NN;
#pragma unroll
  for (int n = 0; n < NN; ++n) { Pws[o + n] = P[n]; Hloc[o + n] = h[n]; }
}

__global__ void scan_phase2_kernel(const float* __restrict__ Pws,
                                   const float* __restrict__ Hloc,
                                   float* __restrict__ Hin) {
  int idx = blockIdx.x * blockDim.x + threadIdx.x;
  if (idx >= BB * DD * NN) return;
  const int n = idx % NN;
  const int d = (idx / NN) % DD;
  const int b = idx / (NN * DD);
  float h = 0.f;
  for (int c = 0; c < NCH; ++c) {
    const size_t o = (size_t)((b * DD + d) * NCH + c) * NN + n;
    Hin[o] = h;
    h = Hloc[o] + Pws[o] * h;
  }
}

__global__ void scan_phase3_kernel(const float* __restrict__ dt,
                                   const float* __restrict__ dtBC,
                                   const float* __restrict__ xc,
                                   const float* __restrict__ A,
                                   const float* __restrict__ Hin,
                                   float* __restrict__ y) {
  int idx = blockIdx.x * blockDim.x + threadIdx.x;
  if (idx >= BB * DD * NCH) return;
  const int d = idx % DD;
  const int c = (idx / DD) % NCH;
  const int b = idx / (DD * NCH);
  float h[NN], Ad[NN];
  const size_t o = (size_t)((b * DD + d) * NCH + c) * NN;
#pragma unroll
  for (int n = 0; n < NN; ++n) { h[n] = Hin[o + n]; Ad[n] = A[d * NN + n]; }
  const int t0 = c * TCH;
  for (int t = t0; t < t0 + TCH; ++t) {
    const size_t row = (size_t)b * SS + t;
    const float dtv = dt[row * DD + d];
    const float xv  = xc[row * DD + d];
    const float* Bt = dtBC + row * (2 * NN) + NN;
    float ysum = 0.f;
#pragma unroll
    for (int n = 0; n < NN; ++n) {
      const float e = __expf(Ad[n] * dtv);
      h[n] = h[n] * e + Bt[n] * xv;
      ysum += h[n];
    }
    y[row * DD + d] = ysum;
  }
}

// Gate: yg = y * silu(z), converted to bf16 for the final GEMM.
__global__ void gate_kernel(const float* __restrict__ y,
                            const float* __restrict__ xz,
                            uint16_t* __restrict__ ygb) {
  int i = blockIdx.x * blockDim.x + threadIdx.x;
  if (i >= MM * DD) return;
  const int d = i % DD;
  const size_t m = (size_t)(i / DD);
  const float z = xz[m * TWO_D + DD + d];
  const float g = z / (1.f + __expf(-z));
  ygb[i] = f2bf(y[i] * g);
}

// ---------------------------------------------------------------- launcher
extern "C" void kernel_launch(void* const* d_in, const int* in_sizes, int n_in,
                              void* d_out, int out_size, void* d_ws, size_t ws_size,
                              hipStream_t stream) {
  (void)in_sizes; (void)n_in; (void)out_size; (void)ws_size;
  const float* x      = (const float*)d_in[0];
  const float* W_in   = (const float*)d_in[1];
  const float* conv_w = (const float*)d_in[2];
  const float* W_x    = (const float*)d_in[3];
  const float* W_dt   = (const float*)d_in[4];
  const float* b_dt   = (const float*)d_in[5];
  const float* A_log  = (const float*)d_in[6];
  const float* W_out  = (const float*)d_in[7];
  float* out = (float*)d_out;

  size_t cur = 0;
  auto take = [&](size_t bytes) -> void* {
    cur = (cur + 255) & ~(size_t)255;
    void* p = (char*)d_ws + cur;
    cur += bytes;
    return p;
  };
  uint16_t* x_bf    = (uint16_t*)take((size_t)MM * DD * 2);
  uint16_t* Win_bf  = (uint16_t*)take((size_t)TWO_D * DD * 2);
  uint16_t* Wx_bf   = (uint16_t*)take((size_t)(2 * NN) * DD * 2);
  uint16_t* Wout_bf = (uint16_t*)take((size_t)DD * DD * 2);
  float*    Aneg    = (float*)take((size_t)DD * NN * 4);
  float*    xz      = (float*)take((size_t)MM * TWO_D * 4);
  float*    xc      = (float*)take((size_t)MM * DD * 4);
  uint16_t* xc_bf   = (uint16_t*)take((size_t)MM * DD * 2);
  float*    dtBC    = (float*)take((size_t)MM * 2 * NN * 4);
  float*    dt      = (float*)take((size_t)MM * DD * 4);
  float*    Pws     = (float*)take((size_t)BB * DD * NCH * NN * 4);
  float*    Hloc    = (float*)take((size_t)BB * DD * NCH * NN * 4);
  float*    Hin     = (float*)take((size_t)BB * DD * NCH * NN * 4);
  float*    y       = (float*)take((size_t)MM * DD * 4);
  uint16_t* yg_bf   = (uint16_t*)take((size_t)MM * DD * 2);

  const int TB = 256;
  auto nb = [](int n, int tb) { return (n + tb - 1) / tb; };

  // operand precision conversion + A = -exp(A_log)
  cvt_bf16x4_kernel<<<nb(MM * DD / 4, TB), TB, 0, stream>>>(x, x_bf, MM * DD / 4);
  cvt_bf16x4_kernel<<<nb(TWO_D * DD / 4, TB), TB, 0, stream>>>(W_in, Win_bf, TWO_D * DD / 4);
  cvt_bf16x4_kernel<<<nb(2 * NN * DD / 4, TB), TB, 0, stream>>>(W_x, Wx_bf, 2 * NN * DD / 4);
  cvt_bf16x4_kernel<<<nb(DD * DD / 4, TB), TB, 0, stream>>>(W_out, Wout_bf, DD * DD / 4);
  negexp_kernel<<<nb(DD * NN, TB), TB, 0, stream>>>(A_log, Aneg, DD * NN);

  // GEMM1: xz = x @ W_in^T   [8192 x 2048] K=1024  (1024 blocks x 128x128 tile)
  gemm_lds_bf16_wmma<<<(MM / 128) * (TWO_D / 128), 256, 0, stream>>>(
      x_bf, Win_bf, xz, MM, TWO_D, DD);

  conv_silu_kernel<<<nb(MM * DD, TB), TB, 0, stream>>>(xz, conv_w, xc, xc_bf);

  // GEMM2: dtBC = xc @ W_x^T [8192 x 32] K=1024  (skinny: direct wave-tile)
  { int waves = (MM / 16) * ((2 * NN) / 32);
    gemm_bf16_wmma<2><<<waves / 8, 256, 0, stream>>>(xc_bf, Wx_bf, dtBC, MM, 2 * NN, DD); }

  dtproj_softplus_kernel<<<nb(MM * DD, TB), TB, 0, stream>>>(dtBC, W_dt, b_dt, dt);

  // chunked parallel linear scan
  scan_phase1_kernel<<<nb(BB * DD * NCH, TB), TB, 0, stream>>>(dt, dtBC, xc, Aneg, Pws, Hloc);
  scan_phase2_kernel<<<nb(BB * DD * NN, TB), TB, 0, stream>>>(Pws, Hloc, Hin);
  scan_phase3_kernel<<<nb(BB * DD * NCH, TB), TB, 0, stream>>>(dt, dtBC, xc, Aneg, Hin, y);

  gate_kernel<<<nb(MM * DD, TB), TB, 0, stream>>>(y, xz, yg_bf);

  // GEMM3: out = yg @ W_out^T [8192 x 1024] K=1024  (512 blocks x 128x128 tile)
  gemm_lds_bf16_wmma<<<(MM / 128) * (DD / 128), 256, 0, stream>>>(
      yg_bf, Wout_bf, out, MM, DD, DD);
}
